// SelectiveSSM_61125974557355
// MI455X (gfx1250) — matrix operand order repaired and checked by reading, not compiled, not run on hardware
//
#include <hip/hip_runtime.h>
#include <hip/hip_bf16.h>

// ---------------------------------------------------------------------------
// Selective SSM (Mamba block) for MI455X / gfx1250, wave32 + WMMA.
//   L = 2048 (seq), D = 2048 (d_model), R = 128 (dt_rank), N = 16 (d_state)
// Pipeline:
//   k0  : f32 -> bf16 pre-conversion of x, W_x, W_dt   (one-time, ~9 MB)
//   k1  : x_dbl = x @ W_x^T   bf16 WMMA; one wave per 16-row band computes all
//         10 column tiles -> A fragment loaded once, x read exactly once.
//   k2  : dt = softplus(dt_in @ W_dt^T + b_dt); wave computes 16x128 strip.
//   k3  : fused selective scan (no dA/dB_x materialization -> avoids 512 MB);
//         chunked LDS staging via GLOBAL_LOAD_ASYNC_TO_LDS when available.
// ---------------------------------------------------------------------------

#define L_SEQ   2048
#define DMODEL  2048
#define RDT     128
#define NST     16
#define XCOLS   (RDT + 2 * NST)   // 160
#define CHUNK   64                // scan LDS chunk (l-steps)

typedef __attribute__((ext_vector_type(16))) __bf16         v16bf;
typedef __attribute__((ext_vector_type(8)))  float          v8f;
typedef __attribute__((ext_vector_type(16))) unsigned short v16u;

union BF16Frag {
    v16u  u;
    uint4 q[2];
    __device__ v16bf bf() const { return __builtin_bit_cast(v16bf, u); }
};

__device__ __forceinline__ unsigned short f32_to_bf16(float f) {
    unsigned u = __builtin_bit_cast(unsigned, f);
    u += 0x7FFFu + ((u >> 16) & 1u);          // round-to-nearest-even
    return (unsigned short)(u >> 16);
}

__device__ __forceinline__ float softplusf(float v) {
    return v > 20.0f ? v : __logf(1.0f + __expf(v));
}

// ---- async global->LDS staging (gfx1250 GLOBAL_LOAD_ASYNC_TO_LDS_B32) -----
#if defined(__HIP_DEVICE_COMPILE__)
#if __has_builtin(__builtin_amdgcn_global_load_async_to_lds_b32)
#define HAVE_ASYNC_LDS 1
#endif
#endif

#ifdef HAVE_ASYNC_LDS
typedef __attribute__((address_space(1))) int as1_int;   // global
typedef __attribute__((address_space(3))) int as3_int;   // LDS
#endif

__device__ __forceinline__ void stage_b32(const float* __restrict__ g, float* l) {
#ifdef HAVE_ASYNC_LDS
    __builtin_amdgcn_global_load_async_to_lds_b32((as1_int*)g, (as3_int*)l, 0, 0);
#else
    *l = *g;
#endif
}

__device__ __forceinline__ void stage_wait() {
#ifdef HAVE_ASYNC_LDS
#if __has_builtin(__builtin_amdgcn_s_wait_asynccnt)
    __builtin_amdgcn_s_wait_asynccnt(0);
#else
    asm volatile("s_wait_asynccnt 0" ::: "memory");
#endif
#endif
}

// ---------------------------------------------------------------------------
// k0: elementwise f32 -> bf16
// ---------------------------------------------------------------------------
__global__ void ssm_cvt_bf16(const float* __restrict__ in,
                             unsigned short* __restrict__ out, int n) {
    int i = blockIdx.x * blockDim.x + threadIdx.x;
    if (i < n) out[i] = f32_to_bf16(in[i]);
}

// ---------------------------------------------------------------------------
// k1: x_dbl[L,160] = x[L,2048] @ W_x^T   (W_x row-major [160,2048])
//     One wave per 16-row band; A fragment reused across all 10 N-tiles.
//     Per K-step: 2 A loads + 20 B loads + 10 WMMA  (vs 8 loads / 1 WMMA).
//     Also writes bf16 copy of cols [0,128) as dt_in for k2.
// ---------------------------------------------------------------------------
__global__ void ssm_gemm_xdbl(const unsigned short* __restrict__ xb,
                              const unsigned short* __restrict__ wxb,
                              float*          __restrict__ xdbl,
                              unsigned short* __restrict__ dtinb) {
    const int wave = threadIdx.x >> 5;
    const int lane = threadIdx.x & 31;
    const int tm   = blockIdx.x * (blockDim.x >> 5) + wave;    // 0..127 row band
    const int half = lane >> 4;
    const int id16 = lane & 15;

    // A: lanes 0-15 hold rows M=0..15; half selects K sub-range
    const unsigned short* aPtr = xb  + (size_t)(tm * 16 + id16) * DMODEL;
    // B: lane holds column (t*16 + id16); per-t displacement folds to imm offset
    const unsigned short* bPtr = wxb + (size_t)id16 * DMODEL;

    v8f acc[XCOLS / 16];
#pragma unroll
    for (int t = 0; t < XCOLS / 16; ++t) acc[t] = (v8f){};

    for (int kb = 0; kb < DMODEL; kb += 32) {
        BF16Frag a;
        a.q[0] = *reinterpret_cast<const uint4*>(aPtr + kb + half * 8);
        a.q[1] = *reinterpret_cast<const uint4*>(aPtr + kb + 16 + half * 8);
#pragma unroll
        for (int t = 0; t < XCOLS / 16; ++t) {
            const unsigned short* bt = bPtr + (size_t)t * 16 * DMODEL + kb + half * 8;
            BF16Frag b;
            b.q[0] = *reinterpret_cast<const uint4*>(bt);
            b.q[1] = *reinterpret_cast<const uint4*>(bt + 16);
            acc[t] = __builtin_amdgcn_wmma_f32_16x16x32_bf16(
                         false, a.bf(), false, b.bf(), (short)0, acc[t], false, false);
        }
    }

    // D layout: VGPR r -> row (tm*16 + half*8 + r), col (t*16 + id16)
    const int rowBase = tm * 16 + half * 8;
#pragma unroll
    for (int t = 0; t < XCOLS / 16; ++t) {
        const int col = t * 16 + id16;
        float* outf = xdbl + (size_t)rowBase * XCOLS + col;
#pragma unroll
        for (int r = 0; r < 8; ++r) outf[(size_t)r * XCOLS] = acc[t][r];
        if (t < RDT / 16) {                       // dt_in columns, uniform branch
            unsigned short* outb = dtinb + (size_t)rowBase * RDT + col;
#pragma unroll
            for (int r = 0; r < 8; ++r) outb[(size_t)r * RDT] = f32_to_bf16(acc[t][r]);
        }
    }
}

// ---------------------------------------------------------------------------
// k2: dt[L,2048] = softplus(dt_in[L,128] @ W_dt^T + b_dt)
//     One wave per 16x128 output strip (8 N-tiles, A reused), K = 128.
// ---------------------------------------------------------------------------
__global__ void ssm_gemm_dt(const unsigned short* __restrict__ dtinb,
                            const unsigned short* __restrict__ wdtb,
                            const float* __restrict__ b_dt,
                            float* __restrict__ dt) {
    const int wave = threadIdx.x >> 5;
    const int lane = threadIdx.x & 31;
    const int wid  = blockIdx.x * (blockDim.x >> 5) + wave;    // 0..2047
    const int tm   = wid >> 4;                                 // 0..127 (L band)
    const int cg   = wid & 15;                                 // 0..15 (col group of 128)
    const int half = lane >> 4;
    const int id16 = lane & 15;

    const unsigned short* aPtr = dtinb + (size_t)(tm * 16 + id16) * RDT;
    const unsigned short* bPtr = wdtb  + (size_t)(cg * 128 + id16) * RDT;

    v8f acc[8];
#pragma unroll
    for (int t = 0; t < 8; ++t) acc[t] = (v8f){};

#pragma unroll
    for (int kb = 0; kb < RDT; kb += 32) {
        BF16Frag a;
        a.q[0] = *reinterpret_cast<const uint4*>(aPtr + kb + half * 8);
        a.q[1] = *reinterpret_cast<const uint4*>(aPtr + kb + 16 + half * 8);
#pragma unroll
        for (int t = 0; t < 8; ++t) {
            const unsigned short* bt = bPtr + (size_t)t * 16 * RDT + kb + half * 8;
            BF16Frag b;
            b.q[0] = *reinterpret_cast<const uint4*>(bt);
            b.q[1] = *reinterpret_cast<const uint4*>(bt + 16);
            acc[t] = __builtin_amdgcn_wmma_f32_16x16x32_bf16(
                         false, a.bf(), false, b.bf(), (short)0, acc[t], false, false);
        }
    }

    const int rowBase = tm * 16 + half * 8;
#pragma unroll
    for (int t = 0; t < 8; ++t) {
        const int   col  = cg * 128 + t * 16 + id16;
        const float bias = b_dt[col];
        float* out = dt + (size_t)rowBase * DMODEL + col;
#pragma unroll
        for (int r = 0; r < 8; ++r)
            out[(size_t)r * DMODEL] = softplusf(acc[t][r] + bias);
    }
}

// ---------------------------------------------------------------------------
// k3: fused selective scan.
//     16 lanes (n=0..15) = the 16 states of one d-channel; 2 channels/wave;
//     block = 256 threads = 16 d-channels; 128 blocks cover D=2048.
//     h = exp(dt*A)*h + dt*B*x  (inclusive scan), y = sum_n h*C + x*D.
// ---------------------------------------------------------------------------
__global__ void ssm_scan(const float* __restrict__ x,
                         const float* __restrict__ xdbl,
                         const float* __restrict__ dt,
                         const float* __restrict__ A_log,
                         const float* __restrict__ Dparam,
                         float* __restrict__ y) {
    __shared__ float s_dt[CHUNK][16];
    __shared__ float s_x [CHUNK][16];
    __shared__ float s_bc[CHUNK][32];   // cols 128..159 of x_dbl: B | C

    const int tid = threadIdx.x;
    const int g   = tid >> 4;                 // channel within block (0..15)
    const int n   = tid & 15;                 // state index
    const int d0  = blockIdx.x * 16;
    const int d   = d0 + g;

    const float a  = -__expf(A_log[(size_t)d * NST + n]);
    const float Dd = Dparam[d];
    float h = 0.0f;

    for (int lc = 0; lc < L_SEQ; lc += CHUNK) {
        __syncthreads();
        // stage chunk: dt/x tiles [CHUNK][16], B|C tile [CHUNK][32]
        for (int r = tid >> 4; r < CHUNK; r += 16) {
            stage_b32(&dt[(size_t)(lc + r) * DMODEL + d0 + n], &s_dt[r][n]);
            stage_b32(&x [(size_t)(lc + r) * DMODEL + d0 + n], &s_x [r][n]);
        }
        {
            const int c = tid & 31;
            for (int r = tid >> 5; r < CHUNK; r += 8)
                stage_b32(&xdbl[(size_t)(lc + r) * XCOLS + RDT + c], &s_bc[r][c]);
        }
        stage_wait();                      // drain this wave's ASYNCcnt
        __syncthreads();                   // publish LDS across waves

        for (int l = 0; l < CHUNK; ++l) {
            const float dtv = s_dt[l][g];
            const float xv  = s_x [l][g];
            const float bv  = s_bc[l][n];
            const float cv  = s_bc[l][16 + n];
            h = __expf(dtv * a) * h + dtv * bv * xv;
            float p = h * cv;
            p += __shfl_xor(p, 1);            // butterfly over the 16 states
            p += __shfl_xor(p, 2);            // (xor<16 stays within half-wave)
            p += __shfl_xor(p, 4);
            p += __shfl_xor(p, 8);
            if (n == 0)
                y[(size_t)(lc + l) * DMODEL + d] = p + xv * Dd;
        }
    }
}

// ---------------------------------------------------------------------------
// Workspace layout (bytes), all 256B aligned. Total ~27 MB.
// ---------------------------------------------------------------------------
#define OFF_XB    ((size_t)0)                       // x bf16     [L,2048]  8 MB
#define OFF_WXB   ((size_t)8388608)                 // W_x bf16   [160,2048] 640 KB
#define OFF_WDTB  ((size_t)9043968)                 // W_dt bf16  [2048,128] 512 KB
#define OFF_DTINB ((size_t)9568256)                 // dt_in bf16 [L,128]   512 KB
#define OFF_XDBL  ((size_t)10092544)                // x_dbl f32  [L,160]   1.25 MB
#define OFF_DT    ((size_t)11403264)                // dt f32     [L,2048]  16 MB

extern "C" void kernel_launch(void* const* d_in, const int* in_sizes, int n_in,
                              void* d_out, int out_size, void* d_ws, size_t ws_size,
                              hipStream_t stream) {
    const float* x     = (const float*)d_in[0];   // [2048,2048]
    const float* W_x   = (const float*)d_in[1];   // [160,2048]
    const float* W_dt  = (const float*)d_in[2];   // [2048,128]
    const float* b_dt  = (const float*)d_in[3];   // [2048]
    const float* A_log = (const float*)d_in[4];   // [2048,16]
    const float* Dp    = (const float*)d_in[5];   // [2048]
    float*       y     = (float*)d_out;           // [2048,2048]

    char* ws = (char*)d_ws;
    unsigned short* xb    = (unsigned short*)(ws + OFF_XB);
    unsigned short* wxb   = (unsigned short*)(ws + OFF_WXB);
    unsigned short* wdtb  = (unsigned short*)(ws + OFF_WDTB);
    unsigned short* dtinb = (unsigned short*)(ws + OFF_DTINB);
    float*          xdbl  = (float*)(ws + OFF_XDBL);
    float*          dtbuf = (float*)(ws + OFF_DT);

    // k0: pre-convert to bf16
    ssm_cvt_bf16<<<(L_SEQ * DMODEL + 255) / 256, 256, 0, stream>>>(x, xb, L_SEQ * DMODEL);
    ssm_cvt_bf16<<<(XCOLS * DMODEL + 255) / 256, 256, 0, stream>>>(W_x, wxb, XCOLS * DMODEL);
    ssm_cvt_bf16<<<(DMODEL * RDT + 255) / 256, 256, 0, stream>>>(W_dt, wdtb, DMODEL * RDT);

    // k1: 128 row-band waves, 8 waves/block -> 16 blocks
    ssm_gemm_xdbl<<<16, 256, 0, stream>>>(xb, wxb, xdbl, dtinb);

    // k2: 2048 strip waves, 8 waves/block -> 256 blocks
    ssm_gemm_dt<<<256, 256, 0, stream>>>(dtinb, wdtb, b_dt, dtbuf);

    // k3: 128 blocks x 256 threads (16 d-channels each)
    ssm_scan<<<DMODEL / 16, 256, 0, stream>>>(x, xdbl, dtbuf, A_log, Dp, y);
}
